// SparseTopKMoE_20633022890187
// MI455X (gfx1250) — matrix operand order repaired and checked by reading, not compiled, not run on hardware
//
#include <hip/hip_runtime.h>
#include <hip/hip_bf16.h>

// ---------------------------------------------------------------------------
// SparseTopKMoE forward for MI455X (gfx1250), wave32, WMMA bf16 tensor cores.
// GEMMs/attention run on v_wmma_f32_16x16x32_bf16 with fp32 accumulation.
// LDS tiles are laid out so every fragment load is 2x ds_load_b128:
//   As: [m][k] row-major (A frag = two contiguous 8-element K runs)
//   Bs: [n][k] transposed (B frag = two contiguous 8-element K runs)
// Global->LDS staging is float4-vectorized and double-buffered via registers.
// fp32->bf16 uses the native __bf16 cast (fptrunc, RNE) so the backend can
// select hardware bf16 converts on gfx1250.
// ---------------------------------------------------------------------------

typedef __attribute__((ext_vector_type(16))) __bf16   v16bf;
typedef __attribute__((ext_vector_type(2)))  __bf16   v2bf;
typedef __attribute__((ext_vector_type(8)))  float    v8f;

#define B_    2
#define S_    1024
#define H_    512
#define NH_   8
#define DH_   64
#define FF_   2048
#define V_    32000
#define E_    3
#define NTOK  (B_ * S_)   // 2048
#define H3_   (3 * H_)    // 1536

// ---------------- bf16 helpers (native fptrunc, RNE) -----------------------
__device__ inline __bf16 f2bf(float f) { return static_cast<__bf16>(f); }
__device__ inline unsigned short f2bfu(float f) {
  return __builtin_bit_cast(unsigned short, static_cast<__bf16>(f));
}
__device__ inline unsigned pack2bf(float lo, float hi) {
  v2bf r;
  r[0] = static_cast<__bf16>(lo);
  r[1] = static_cast<__bf16>(hi);
  return __builtin_bit_cast(unsigned, r);
}
__device__ inline __bf16 u2bf(unsigned short s) {
  return __builtin_bit_cast(__bf16, s);
}
// Convert 16 fp32 (four float4s, in order) into a v16bf fragment.
__device__ inline v16bf pack16(float4 a, float4 b, float4 c, float4 d) {
  v16bf r;
  r[0]  = f2bf(a.x); r[1]  = f2bf(a.y); r[2]  = f2bf(a.z); r[3]  = f2bf(a.w);
  r[4]  = f2bf(b.x); r[5]  = f2bf(b.y); r[6]  = f2bf(b.z); r[7]  = f2bf(b.w);
  r[8]  = f2bf(c.x); r[9]  = f2bf(c.y); r[10] = f2bf(c.z); r[11] = f2bf(c.w);
  r[12] = f2bf(d.x); r[13] = f2bf(d.y); r[14] = f2bf(d.z); r[15] = f2bf(d.w);
  return r;
}

__device__ inline float wave_max(float x) {
#pragma unroll
  for (int off = 16; off > 0; off >>= 1) x = fmaxf(x, __shfl_xor(x, off, 32));
  return x;
}
__device__ inline float wave_sum(float x) {
#pragma unroll
  for (int off = 16; off > 0; off >>= 1) x += __shfl_xor(x, off, 32);
  return x;
}
__device__ inline float gelu_tanh(float x) {
  float x3 = x * x * x;
  return 0.5f * x * (1.0f + tanhf(0.7978845608028654f * (x + 0.044715f * x3)));
}

#define BM 128
#define BN 128
#define BK 32

// ---------------- WMMA fragment loaders (per CDNA5 ISA VGPR layout) --------
// A 16x32 bf16 (MxK), LDS row-major [m][BK]:
//  lane L: m = m0 + (L&15), g = L>>4; elem e -> k = ((e>>3)<<4) + g*8 + (e&7)
//  => two contiguous 8-elem runs at k = g*8 and k = 16 + g*8 (2x b128)
__device__ inline v16bf load_a_frag_lds(const unsigned short* As, int m0, int ld) {
  int lane = threadIdx.x & 31;
  int grp  = lane >> 4;
  const unsigned short* p = As + (m0 + (lane & 15)) * ld + grp * 8;
  v16bf a;
#pragma unroll
  for (int e = 0; e < 8; ++e)  a[e]     = u2bf(p[e]);
#pragma unroll
  for (int e = 0; e < 8; ++e)  a[8 + e] = u2bf(p[16 + e]);
  return a;
}
// B 32x16 bf16 (KxN), LDS transposed [n][ldk]:
//  lane L: n = n0 + (L&15), g = L>>4; elem e -> k = ((e>>3)<<4) + g*8 + (e&7)
//  => two contiguous 8-elem runs at k = g*8 and k = 16 + g*8 (2x b128)
__device__ inline v16bf load_b_frag_t(const unsigned short* Bs, int n0, int ldk) {
  int lane = threadIdx.x & 31;
  int grp  = lane >> 4;
  const unsigned short* p = Bs + (n0 + (lane & 15)) * ldk + grp * 8;
  v16bf b;
#pragma unroll
  for (int e = 0; e < 8; ++e)  b[e]     = u2bf(p[e]);
#pragma unroll
  for (int e = 0; e < 8; ++e)  b[8 + e] = u2bf(p[16 + e]);
  return b;
}

// ---------------------------------------------------------------------------
// Generic GEMM: C[M,N] = act(A[M,K] @ W[K,N] + bias).
// Block 256 threads (8 waves), tile 128x128x32, wave tile 64x32.
// Register-prefetch double buffering of global->LDS staging.
// ---------------------------------------------------------------------------
__global__ void __launch_bounds__(256)
gemm_kernel(const float* __restrict__ A, const float* __restrict__ W,
            const float* __restrict__ bias, float* __restrict__ C,
            int M, int N, int K, int act) {
  __shared__ unsigned short As[BM * BK];   // [m][k]
  __shared__ unsigned short Bs[BN * BK];   // [n][k] (transposed)
  unsigned int* As32 = (unsigned int*)As;

  int tid  = threadIdx.x;
  int wave = tid >> 5;
  int m_wave = (wave >> 2) * 64;
  int n_wave = (wave & 3) * 32;
  int blk_m = blockIdx.y * BM;
  int blk_n = blockIdx.x * BN;

  v8f acc[4][2];
#pragma unroll
  for (int i = 0; i < 4; ++i)
#pragma unroll
    for (int j = 0; j < 2; ++j)
#pragma unroll
      for (int v = 0; v < 8; ++v) acc[i][j][v] = 0.0f;

  float4 ra[4], rb[4];
  auto load_tiles = [&](int kt) {
#pragma unroll
    for (int it = 0; it < 4; ++it) {
      int i = tid + it * 256;                 // 0..1023
      int m = i >> 3, kq = (i & 7) << 2;      // A: 8 quads per row
      ra[it] = *(const float4*)(A + (size_t)(blk_m + m) * K + kt + kq);
      int k = i >> 5, nq = (i & 31) << 2;     // B: 32 quads per k-row
      rb[it] = *(const float4*)(W + (size_t)(kt + k) * N + blk_n + nq);
    }
  };
  auto store_tiles = [&]() {
#pragma unroll
    for (int it = 0; it < 4; ++it) {
      int i = tid + it * 256;
      int m = i >> 3, kq = (i & 7) << 2;
      int idx = (m * BK + kq) >> 1;
      As32[idx]     = pack2bf(ra[it].x, ra[it].y);
      As32[idx + 1] = pack2bf(ra[it].z, ra[it].w);
      int k = i >> 5, nq = (i & 31) << 2;
      Bs[(nq + 0) * BK + k] = f2bfu(rb[it].x);
      Bs[(nq + 1) * BK + k] = f2bfu(rb[it].y);
      Bs[(nq + 2) * BK + k] = f2bfu(rb[it].z);
      Bs[(nq + 3) * BK + k] = f2bfu(rb[it].w);
    }
  };

  load_tiles(0);
  for (int kt = 0; kt < K; kt += BK) {
    store_tiles();
    if (kt + BK < K) load_tiles(kt + BK);     // overlap next-tile loads w/ WMMA
    __syncthreads();

    v16bf bfrag[2];
#pragma unroll
    for (int j = 0; j < 2; ++j) bfrag[j] = load_b_frag_t(Bs, n_wave + j * 16, BK);
#pragma unroll
    for (int i = 0; i < 4; ++i) {
      v16bf afrag = load_a_frag_lds(As, m_wave + i * 16, BK);
#pragma unroll
      for (int j = 0; j < 2; ++j)
        acc[i][j] = __builtin_amdgcn_wmma_f32_16x16x32_bf16(
            false, afrag, false, bfrag[j], (short)0, acc[i][j], false, false);
    }
    __syncthreads();
  }

  int lane = tid & 31, grp = lane >> 4, nl = lane & 15;
#pragma unroll
  for (int i = 0; i < 4; ++i)
#pragma unroll
    for (int j = 0; j < 2; ++j)
#pragma unroll
      for (int v = 0; v < 8; ++v) {
        int m = blk_m + m_wave + i * 16 + v + 8 * grp;
        int n = blk_n + n_wave + j * 16 + nl;
        float x = acc[i][j][v] + (bias ? bias[n] : 0.0f);
        if (act == 1) x = gelu_tanh(x);
        C[(size_t)m * N + n] = x;
      }
}

// ---------------------------------------------------------------------------
// Vocab projection + top-k combine:
// out[b,s,:] = sum_e gates[b,e] * (lnout[e,b,s,:] @ embed_out[e])
// Gate is folded into the bf16 A-tile conversion.  grid = (V/128, S/128, B).
// ---------------------------------------------------------------------------
__global__ void __launch_bounds__(256)
vocab_kernel(const float* __restrict__ lnout, const float* __restrict__ wout,
             const float* __restrict__ gates, float* __restrict__ out) {
  __shared__ unsigned short As[BM * BK];
  __shared__ unsigned short Bs[BN * BK];
  unsigned int* As32 = (unsigned int*)As;

  int tid  = threadIdx.x;
  int wave = tid >> 5;
  int m_wave = (wave >> 2) * 64;
  int n_wave = (wave & 3) * 32;
  int blk_m = blockIdx.y * BM;
  int blk_n = blockIdx.x * BN;
  int b     = blockIdx.z;

  v8f acc[4][2];
#pragma unroll
  for (int i = 0; i < 4; ++i)
#pragma unroll
    for (int j = 0; j < 2; ++j)
#pragma unroll
      for (int v = 0; v < 8; ++v) acc[i][j][v] = 0.0f;

  for (int e = 0; e < E_; ++e) {
    float gate = gates[b * E_ + e];
    if (gate == 0.0f) continue;
    const float* A = lnout + ((size_t)e * NTOK + (size_t)b * S_) * H_;  // [S,H]
    const float* W = wout + (size_t)e * H_ * V_;                         // [H,V]

    float4 ra[4], rb[4];
    auto load_tiles = [&](int kt) {
#pragma unroll
      for (int it = 0; it < 4; ++it) {
        int i = tid + it * 256;
        int m = i >> 3, kq = (i & 7) << 2;
        ra[it] = *(const float4*)(A + (size_t)(blk_m + m) * H_ + kt + kq);
        int k = i >> 5, nq = (i & 31) << 2;
        rb[it] = *(const float4*)(W + (size_t)(kt + k) * V_ + blk_n + nq);
      }
    };
    auto store_tiles = [&]() {
#pragma unroll
      for (int it = 0; it < 4; ++it) {
        int i = tid + it * 256;
        int m = i >> 3, kq = (i & 7) << 2;
        int idx = (m * BK + kq) >> 1;
        As32[idx]     = pack2bf(gate * ra[it].x, gate * ra[it].y);
        As32[idx + 1] = pack2bf(gate * ra[it].z, gate * ra[it].w);
        int k = i >> 5, nq = (i & 31) << 2;
        Bs[(nq + 0) * BK + k] = f2bfu(rb[it].x);
        Bs[(nq + 1) * BK + k] = f2bfu(rb[it].y);
        Bs[(nq + 2) * BK + k] = f2bfu(rb[it].z);
        Bs[(nq + 3) * BK + k] = f2bfu(rb[it].w);
      }
    };

    load_tiles(0);
    for (int kt = 0; kt < H_; kt += BK) {
      store_tiles();
      if (kt + BK < H_) load_tiles(kt + BK);
      __syncthreads();

      v16bf bfrag[2];
#pragma unroll
      for (int j = 0; j < 2; ++j) bfrag[j] = load_b_frag_t(Bs, n_wave + j * 16, BK);
#pragma unroll
      for (int i = 0; i < 4; ++i) {
        v16bf afrag = load_a_frag_lds(As, m_wave + i * 16, BK);
#pragma unroll
        for (int j = 0; j < 2; ++j)
          acc[i][j] = __builtin_amdgcn_wmma_f32_16x16x32_bf16(
              false, afrag, false, bfrag[j], (short)0, acc[i][j], false, false);
      }
      __syncthreads();
    }
  }

  int lane = tid & 31, grp = lane >> 4, nl = lane & 15;
#pragma unroll
  for (int i = 0; i < 4; ++i)
#pragma unroll
    for (int j = 0; j < 2; ++j)
#pragma unroll
      for (int v = 0; v < 8; ++v) {
        int m = blk_m + m_wave + i * 16 + v + 8 * grp;   // 0..1023 within b
        int n = blk_n + n_wave + j * 16 + nl;
        out[((size_t)b * S_ + m) * V_ + n] = acc[i][j][v];
      }
}

// ---------------------------------------------------------------------------
// Causal multi-head attention.  One block = (b, head, 64-query tile).
// 128 threads (4 waves).  Dynamic LDS: full score rows 64 x 1024 f32 (256KB)
// + transposed V chunk 64x32 bf16 (4KB) — legal in CDNA5's 320KB/WGP.
// ---------------------------------------------------------------------------
__global__ void __launch_bounds__(128)
attn_kernel(const float* __restrict__ qkv, float* __restrict__ ctx) {
  extern __shared__ float sc[];                          // [64][S_]
  unsigned short* Vs = (unsigned short*)(sc + 64 * S_);  // [dh=64][key=32]

  int bid = blockIdx.x;
  int qt  = bid & 15;            // S_/64 tiles
  int hd  = (bid >> 4) & (NH_ - 1);
  int b   = bid >> 7;

  int tid  = threadIdx.x;
  int wave = tid >> 5;
  int lane = tid & 31;
  int grp  = lane >> 4;
  int l16  = lane & 15;

  const float* qbase = qkv + (size_t)b * S_ * H3_;
  int qm0 = qt * 64 + wave * 16;   // first global query row for this wave

  // ---- Phase 1: scores = (Q K^T) / sqrt(DH) into LDS -----------------------
  v16bf qf[2];
#pragma unroll
  for (int kk = 0; kk < 2; ++kk) {
    const float* qp =
        qbase + (size_t)(qm0 + l16) * H3_ + hd * DH_ + kk * 32 + grp * 8;
    qf[kk] = pack16(*(const float4*)(qp),      *(const float4*)(qp + 4),
                    *(const float4*)(qp + 16), *(const float4*)(qp + 20));
  }
  for (int nt = 0; nt < S_ / 16; ++nt) {
    if (nt * 16 > qm0 + 15) continue;  // fully masked tile; softmax overwrites
    v8f acc;
#pragma unroll
    for (int v = 0; v < 8; ++v) acc[v] = 0.0f;
#pragma unroll
    for (int kk = 0; kk < 2; ++kk) {
      // B frag of K^T: n = key, elem e -> dh = kk*32 + ((e>>3)<<4) + grp*8 + (e&7)
      const float* kp = qbase + (size_t)(nt * 16 + l16) * H3_ + H_ + hd * DH_ +
                        kk * 32 + grp * 8;
      v16bf bf = pack16(*(const float4*)(kp),      *(const float4*)(kp + 4),
                        *(const float4*)(kp + 16), *(const float4*)(kp + 20));
      acc = __builtin_amdgcn_wmma_f32_16x16x32_bf16(false, qf[kk], false, bf,
                                                    (short)0, acc, false, false);
    }
#pragma unroll
    for (int v = 0; v < 8; ++v) {
      int r = wave * 16 + v + 8 * grp;               // local row 0..63
      sc[r * S_ + nt * 16 + l16] = acc[v] * 0.125f;  // 1/sqrt(64)
    }
  }
  __syncthreads();

  // ---- Phase 2: causal softmax (writes exact zeros into masked columns) ----
  for (int i = 0; i < 16; ++i) {
    int r  = wave + 4 * i;
    int qg = qt * 64 + r;
    float vals[32];
    float mx = -3.0e38f;
#pragma unroll
    for (int j = 0; j < 32; ++j) {
      int c = lane + 32 * j;
      float x = (c <= qg) ? sc[r * S_ + c] : -1.0e9f;
      vals[j] = x;
      mx = fmaxf(mx, x);
    }
    mx = wave_max(mx);
    float sum = 0.0f;
#pragma unroll
    for (int j = 0; j < 32; ++j) {
      vals[j] = expf(vals[j] - mx);
      sum += vals[j];
    }
    sum = wave_sum(sum);
    float inv = 1.0f / sum;
#pragma unroll
    for (int j = 0; j < 32; ++j) sc[r * S_ + lane + 32 * j] = vals[j] * inv;
  }
  __syncthreads();

  // ---- Phase 3: out = P @ V (V chunk staged transposed in LDS) -------------
  v8f o[4];
#pragma unroll
  for (int j = 0; j < 4; ++j)
#pragma unroll
    for (int v = 0; v < 8; ++v) o[j][v] = 0.0f;

  int ktmax_blk = (qt * 64 + 63) >> 5;  // block-uniform bound (P is 0 beyond)
  for (int kt = 0; kt <= ktmax_blk; ++kt) {
    // cooperative stage: Vs[dh][key] for keys kt*32..kt*32+31
    for (int i = tid; i < 512; i += 128) {   // 32 keys * 16 quads
      int key = i >> 4;
      int dq  = (i & 15) << 2;
      const float4 f = *(const float4*)(qbase + (size_t)(kt * 32 + key) * H3_ +
                                        2 * H_ + hd * DH_ + dq);
      Vs[(dq + 0) * 32 + key] = f2bfu(f.x);
      Vs[(dq + 1) * 32 + key] = f2bfu(f.y);
      Vs[(dq + 2) * 32 + key] = f2bfu(f.z);
      Vs[(dq + 3) * 32 + key] = f2bfu(f.w);
    }
    __syncthreads();

    // A frag of P from score rows (f32 LDS, two contiguous 8-float runs)
    const float* pp = sc + (size_t)(wave * 16 + l16) * S_ + kt * 32 + grp * 8;
    v16bf pf = pack16(*(const float4*)(pp),      *(const float4*)(pp + 4),
                      *(const float4*)(pp + 16), *(const float4*)(pp + 20));
#pragma unroll
    for (int j = 0; j < 4; ++j) {
      v16bf vf = load_b_frag_t(Vs, j * 16, 32);
      o[j] = __builtin_amdgcn_wmma_f32_16x16x32_bf16(false, pf, false, vf,
                                                     (short)0, o[j], false, false);
    }
    __syncthreads();
  }
#pragma unroll
  for (int j = 0; j < 4; ++j)
#pragma unroll
    for (int v = 0; v < 8; ++v) {
      int m = qt * 64 + wave * 16 + v + 8 * grp;
      int n = j * 16 + l16;
      ctx[((size_t)b * S_ + m) * H_ + hd * DH_ + n] = o[j][v];
    }
}

// ---------------------------------------------------------------------------
// LayerNorm: one wave per 512-wide row.  grid = NTOK/8, block = 256.
// ---------------------------------------------------------------------------
__global__ void __launch_bounds__(256)
ln_kernel(const float* __restrict__ x, const float* __restrict__ gm,
          const float* __restrict__ bt, float* __restrict__ out, int nrows) {
  int wave = threadIdx.x >> 5, lane = threadIdx.x & 31;
  int row = blockIdx.x * 8 + wave;
  if (row >= nrows) return;
  const float* xr = x + (size_t)row * H_;
  float v[16];
  float s = 0.0f;
#pragma unroll
  for (int i = 0; i < 16; ++i) {
    v[i] = xr[lane + 32 * i];
    s += v[i];
  }
  s = wave_sum(s);
  float mean = s * (1.0f / H_);
  float var = 0.0f;
#pragma unroll
  for (int i = 0; i < 16; ++i) {
    float d = v[i] - mean;
    var += d * d;
  }
  var = wave_sum(var) * (1.0f / H_);
  float inv = rsqrtf(var + 1e-5f);
#pragma unroll
  for (int i = 0; i < 16; ++i) {
    int c = lane + 32 * i;
    out[(size_t)row * H_ + c] = (v[i] - mean) * inv * gm[c] + bt[c];
  }
}

// ---------------------------------------------------------------------------
__global__ void embed_kernel(const int* __restrict__ ids,
                             const float* __restrict__ emb,
                             float* __restrict__ out) {
  int n = blockIdx.x;
  int id = ids[n];
  for (int c = threadIdx.x; c < H_; c += blockDim.x)
    out[(size_t)n * H_ + c] = emb[(size_t)id * H_ + c];
}

__global__ void add3_kernel(float* __restrict__ x, const float* __restrict__ a,
                            const float* __restrict__ m, int n) {
  int i = blockIdx.x * blockDim.x + threadIdx.x;
  if (i < n) x[i] = x[i] + a[i] + m[i];
}

__global__ void copy_kernel(const float* __restrict__ src, float* __restrict__ dst,
                            int n) {
  int i = blockIdx.x * blockDim.x + threadIdx.x;
  if (i < n) dst[i] = src[i];
}

// Mean-pool over sequence, route, softmax, top-2, renormalize.
// gates[b,e] = renormalized weight for selected experts, 0 otherwise.
__global__ void __launch_bounds__(256)
route_kernel(const float* __restrict__ fh, const float* __restrict__ rw,
             float* __restrict__ gates) {
  __shared__ float pooled[H_];
  int b = blockIdx.x;
  for (int c = threadIdx.x; c < H_; c += 256) {
    float s = 0.0f;
    for (int t = 0; t < S_; ++t) s += fh[((size_t)b * S_ + t) * H_ + c];
    pooled[c] = s * (1.0f / S_);
  }
  __syncthreads();
  if (threadIdx.x == 0) {
    float lg[E_] = {0.0f, 0.0f, 0.0f};
    for (int c = 0; c < H_; ++c) {
      float p = pooled[c];
      lg[0] += p * rw[c * E_ + 0];
      lg[1] += p * rw[c * E_ + 1];
      lg[2] += p * rw[c * E_ + 2];
    }
    float mx = fmaxf(lg[0], fmaxf(lg[1], lg[2]));
    float ex[E_], den = 0.0f;
    for (int e = 0; e < E_; ++e) { ex[e] = expf(lg[e] - mx); den += ex[e]; }
    float pv[E_];
    for (int e = 0; e < E_; ++e) pv[e] = ex[e] / den;
    int i0 = 0;
    for (int e = 1; e < E_; ++e) if (pv[e] > pv[i0]) i0 = e;   // ties -> lower idx
    int i1 = -1;
    for (int e = 0; e < E_; ++e)
      if (e != i0 && (i1 < 0 || pv[e] > pv[i1])) i1 = e;
    float wsum = pv[i0] + pv[i1];
    for (int e = 0; e < E_; ++e) gates[b * E_ + e] = 0.0f;
    gates[b * E_ + i0] = pv[i0] / wsum;
    gates[b * E_ + i1] = pv[i1] / wsum;
  }
}

// ---------------------------------------------------------------------------
// Host orchestration
// ---------------------------------------------------------------------------
struct LayerW {
  const float *ln1g, *ln1b, *wqkv, *bqkv, *wo, *bo;
  const float *ln2g, *ln2b, *wfc, *bfc, *wproj, *bproj;
};

static void run_block(const LayerW& L, float* x, float* h, float* qkv, float* ctx,
                      float* attn_out, float* mlpmid, float* mlp_out,
                      hipStream_t stream) {
  int attn_lds = 64 * S_ * (int)sizeof(float) + 64 * 32 * (int)sizeof(unsigned short);
  ln_kernel<<<NTOK / 8, 256, 0, stream>>>(x, L.ln1g, L.ln1b, h, NTOK);
  gemm_kernel<<<dim3(H3_ / BN, NTOK / BM), 256, 0, stream>>>(
      h, L.wqkv, L.bqkv, qkv, NTOK, H3_, H_, 0);
  attn_kernel<<<B_ * NH_ * (S_ / 64), 128, attn_lds, stream>>>(qkv, ctx);
  gemm_kernel<<<dim3(H_ / BN, NTOK / BM), 256, 0, stream>>>(
      ctx, L.wo, L.bo, attn_out, NTOK, H_, H_, 0);
  ln_kernel<<<NTOK / 8, 256, 0, stream>>>(x, L.ln2g, L.ln2b, h, NTOK);
  gemm_kernel<<<dim3(FF_ / BN, NTOK / BM), 256, 0, stream>>>(
      h, L.wfc, L.bfc, mlpmid, NTOK, FF_, H_, 1 /*gelu*/);
  gemm_kernel<<<dim3(H_ / BN, NTOK / BM), 256, 0, stream>>>(
      mlpmid, L.wproj, L.bproj, mlp_out, NTOK, H_, FF_, 0);
  add3_kernel<<<(NTOK * H_) / 256, 256, 0, stream>>>(x, attn_out, mlp_out, NTOK * H_);
}

// d_in flattening (setup_inputs order; inner dicts in sorted-key order):
// 0 input_ids, 1 embed_in,
// frozen: 2 b_fc, 3 b_o, 4 b_proj, 5 b_qkv, 6 ln1_b, 7 ln1_g, 8 ln2_b, 9 ln2_g,
//         10 w_fc, 11 w_o, 12 w_proj, 13 w_qkv
// unfrozen: 14..25 same keys with leading [E,LU] axes
// 26 final_ln_g, 27 final_ln_b, 28 embed_out, 29 router_w
static LayerW make_layer(void* const* d_in, int base, int l) {
  LayerW L;
  const float* b_fc   = (const float*)d_in[base + 0];
  const float* b_o    = (const float*)d_in[base + 1];
  const float* b_proj = (const float*)d_in[base + 2];
  const float* b_qkv  = (const float*)d_in[base + 3];
  const float* ln1_b  = (const float*)d_in[base + 4];
  const float* ln1_g  = (const float*)d_in[base + 5];
  const float* ln2_b  = (const float*)d_in[base + 6];
  const float* ln2_g  = (const float*)d_in[base + 7];
  const float* w_fc   = (const float*)d_in[base + 8];
  const float* w_o    = (const float*)d_in[base + 9];
  const float* w_proj = (const float*)d_in[base + 10];
  const float* w_qkv  = (const float*)d_in[base + 11];
  L.ln1g  = ln1_g + (size_t)l * H_;
  L.ln1b  = ln1_b + (size_t)l * H_;
  L.wqkv  = w_qkv + (size_t)l * H_ * H3_;
  L.bqkv  = b_qkv + (size_t)l * H3_;
  L.wo    = w_o + (size_t)l * H_ * H_;
  L.bo    = b_o + (size_t)l * H_;
  L.ln2g  = ln2_g + (size_t)l * H_;
  L.ln2b  = ln2_b + (size_t)l * H_;
  L.wfc   = w_fc + (size_t)l * H_ * FF_;
  L.bfc   = b_fc + (size_t)l * FF_;
  L.wproj = w_proj + (size_t)l * FF_ * H_;
  L.bproj = b_proj + (size_t)l * H_;
  return L;
}

extern "C" void kernel_launch(void* const* d_in, const int* in_sizes, int n_in,
                              void* d_out, int out_size, void* d_ws, size_t ws_size,
                              hipStream_t stream) {
  (void)in_sizes; (void)n_in; (void)out_size;

  const int*   ids  = (const int*)d_in[0];
  const float* emb  = (const float*)d_in[1];
  const float* flg  = (const float*)d_in[26];
  const float* flb  = (const float*)d_in[27];
  const float* eout = (const float*)d_in[28];
  const float* rw   = (const float*)d_in[29];
  float*       out  = (float*)d_out;

  // Workspace layout (floats)
  float* ws = (float*)d_ws;
  size_t off = 0;
  float* x        = ws + off; off += (size_t)NTOK * H_;    // activations
  float* h        = ws + off; off += (size_t)NTOK * H_;    // ln output
  float* qkv      = ws + off; off += (size_t)NTOK * H3_;
  float* ctx      = ws + off; off += (size_t)NTOK * H_;
  float* attn_out = ws + off; off += (size_t)NTOK * H_;
  float* mlpmid   = ws + off; off += (size_t)NTOK * FF_;
  float* mlp_out  = ws + off; off += (size_t)NTOK * H_;
  float* frozen_h = ws + off; off += (size_t)NTOK * H_;
  float* lnout    = ws + off; off += (size_t)E_ * NTOK * H_;
  float* gates    = ws + off; off += 16;
  if (ws_size < off * sizeof(float)) return;  // deterministic guard

  // Attention needs 260KB dynamic LDS (within CDNA5's 320KB/WGP).
  hipFuncSetAttribute((const void*)attn_kernel,
                      hipFuncAttributeMaxDynamicSharedMemorySize,
                      64 * S_ * (int)sizeof(float) +
                          64 * 32 * (int)sizeof(unsigned short));

  // 1) embedding gather
  embed_kernel<<<NTOK, 256, 0, stream>>>(ids, emb, x);

  // 2) frozen stack (3 layers)
  for (int l = 0; l < 3; ++l) {
    LayerW L = make_layer(d_in, 2, l);
    run_block(L, x, h, qkv, ctx, attn_out, mlpmid, mlp_out, stream);
  }
  copy_kernel<<<(NTOK * H_) / 256, 256, 0, stream>>>(x, frozen_h, NTOK * H_);

  // 3) pool + route + top-2 gates
  route_kernel<<<B_, 256, 0, stream>>>(frozen_h, rw, gates);

  // 4) expert stacks (3 layers each) + final LN
  for (int e = 0; e < E_; ++e) {
    copy_kernel<<<(NTOK * H_) / 256, 256, 0, stream>>>(frozen_h, x, NTOK * H_);
    for (int l = 0; l < 3; ++l) {
      LayerW L = make_layer(d_in, 14, e * 3 + l);
      run_block(L, x, h, qkv, ctx, attn_out, mlpmid, mlp_out, stream);
    }
    ln_kernel<<<NTOK / 8, 256, 0, stream>>>(x, flg + (size_t)e * H_,
                                            flb + (size_t)e * H_,
                                            lnout + (size_t)e * NTOK * H_, NTOK);
  }

  // 5) gated vocab projection + combine (skips zero-gate experts on device)
  vocab_kernel<<<dim3(V_ / BN, S_ / BM, B_), 256, 0, stream>>>(lnout, eout, gates, out);
}
